// EuclideanGraphMatching_90426241450715
// MI455X (gfx1250) — matrix operand order repaired and checked by reading, not compiled, not run on hardware
//
#include <hip/hip_runtime.h>

#define BB   8
#define NN   2048
#define CIN  128
#define COUT 256
#define KK   32
#define BN   (BB * NN)   // 16384 rows
#define SROW 2052        // padded LDS row stride (floats): 8*2052 % 64 == 32 -> no bank conflicts

typedef __attribute__((ext_vector_type(16))) __bf16 v16bf;
typedef __attribute__((ext_vector_type(8)))  __bf16 v8bf;
typedef __attribute__((ext_vector_type(8)))  float  v8f;

// ---------------------------------------------------------------------------
// WMMA helpers (CDNA5, wave32).  D = A(16x32 bf16) * B(32x16 bf16) + C(f32)
// ---------------------------------------------------------------------------
static __device__ __forceinline__ v8f wmma_bf16(v16bf a, v16bf b, v8f c) {
  return __builtin_amdgcn_wmma_f32_16x16x32_bf16(
      /*neg_a=*/false, a, /*neg_b=*/false, b,
      /*c_mod=*/(short)0, c, /*reuse_a=*/false, /*reuse_b=*/false);
}

// A-matrix 16x32 bf16 fragment from a row-major [rows x row_stride] bf16 array.
// Lane L holds row (L&15); lanes 0-15 carry K = kb+0..15,
// lanes 16-31 carry K = kb+16..31  -> two contiguous 16B loads/lane.
static __device__ __forceinline__ v16bf
load_a_frag(const __bf16* __restrict__ base, int row_stride, int lane, int kb) {
  int row = lane & 15;
  int hi  = (lane >> 4) & 1;
  const __bf16* p0 = base + (size_t)row * row_stride + kb + hi * 16;
  v8bf lo = *(const v8bf*)p0;
  v8bf hh = *(const v8bf*)(p0 + 8);
  v16bf a;
#pragma unroll
  for (int i = 0; i < 8; ++i) { a[i] = lo[i]; a[8 + i] = hh[i]; }
  return a;
}

// B-matrix 32x16 bf16 fragment where the *source* array is N-major
// (row n holds all K values): B[k][n] = src[n][k].  Lane L holds column
// (L&15); lanes 0-15 carry K = kb+0..15, lanes 16-31 K = kb+16..31
// -> one contiguous 32B load per lane.
static __device__ __forceinline__ v16bf
load_b_frag(const __bf16* __restrict__ base, int row_stride, int lane, int kb) {
  int n  = lane & 15;
  int hi = (lane >> 4) & 1;
  return *(const v16bf*)(base + (size_t)n * row_stride + kb + hi * 16);
}

// (value, index) ordering used everywhere: higher value wins; equal values ->
// lower index wins (matches jax.lax.top_k tie-breaking).
static __device__ __forceinline__ bool betterf(float a, int ai, float b, int bi) {
  return (a > b) || (a == b && ai < bi);
}

// Sorted insert into a per-lane top-4 candidate list (t0 is best).
static __device__ __forceinline__ void ins4(float v, int vi,
    float& t0, int& i0, float& t1, int& i1,
    float& t2, int& i2, float& t3, int& i3) {
  if (!betterf(v, vi, t3, i3)) return;     // common fast-path reject
  if (betterf(v, vi, t2, i2)) {
    t3 = t2; i3 = i2;
    if (betterf(v, vi, t1, i1)) {
      t2 = t1; i2 = i1;
      if (betterf(v, vi, t0, i0)) { t1 = t0; i1 = i0; t0 = v; i0 = vi; }
      else                        { t1 = v;  i1 = vi; }
    } else                        { t2 = v;  i2 = vi; }
  } else                          { t3 = v;  i3 = vi; }
}

// ---------------------------------------------------------------------------
// Kernel 1: f32 -> bf16 elementwise convert (x_s, x_t)
// ---------------------------------------------------------------------------
__global__ void cvt_bf16_kernel(const float* __restrict__ in,
                                __bf16* __restrict__ out, int n) {
  for (int i = blockIdx.x * blockDim.x + threadIdx.x; i < n;
       i += gridDim.x * blockDim.x)
    out[i] = (__bf16)in[i];
}

// Kernel 2: W_psi [CIN x COUT] f32 -> Wt [COUT x CIN] bf16 (N-major for WMMA B)
__global__ void wtrans_kernel(const float* __restrict__ W,
                              __bf16* __restrict__ Wt) {
  int i = blockIdx.x * blockDim.x + threadIdx.x;
  if (i < CIN * COUT) {
    int k = i / COUT, n = i % COUT;
    Wt[(size_t)n * CIN + k] = (__bf16)W[i];
  }
}

// ---------------------------------------------------------------------------
// Kernel 3: node encoder  H[16384 x 256] = Xbf[16384 x 128] @ W  (bf16 WMMA)
// One workgroup (8 waves) per 16-row block; wave w does column tiles 2w,2w+1.
// ---------------------------------------------------------------------------
__global__ void encode_kernel(const __bf16* __restrict__ Xbf,
                              const __bf16* __restrict__ Wt,
                              __bf16* __restrict__ H) {
  const int lane  = threadIdx.x & 31;
  const int wave  = threadIdx.x >> 5;
  const int rbase = blockIdx.x * 16;

  v16bf A[4];  // 16x128 source tile as 4 K-fragments
#pragma unroll
  for (int f = 0; f < 4; ++f)
    A[f] = load_a_frag(Xbf + (size_t)rbase * CIN, CIN, lane, f * 32);

#pragma unroll
  for (int t = 0; t < 2; ++t) {
    const int ct = wave * 2 + t;            // column tile 0..15 (16 cols each)
    v8f acc = {};
#pragma unroll
    for (int f = 0; f < 4; ++f) {
      v16bf Bf = load_b_frag(Wt + (size_t)ct * 16 * CIN, CIN, lane, f * 32);
      acc = wmma_bf16(A[f], Bf, acc);
    }
    // D layout: VGPR r -> row r (lanes 0-15) / row r+8 (lanes 16-31), col = lane&15
    const int col    = ct * 16 + (lane & 15);
    const int rowoff = (lane >> 4) * 8;
#pragma unroll
    for (int r = 0; r < 8; ++r)
      H[(size_t)(rbase + rowoff + r) * COUT + col] = (__bf16)acc[r];
  }
}

// ---------------------------------------------------------------------------
// Kernel 4: fused  S = Hs @ Ht^T  ->  per-row top-32 (desc, low-idx ties)
//           -> softmax -> (values, indices)
// Grid: (NN/16, BB).  Block: 256 (8 waves).  Dynamic LDS: 16*SROW*4 bytes.
// ---------------------------------------------------------------------------
__global__ void simtopk_kernel(const __bf16* __restrict__ Hs,
                               const __bf16* __restrict__ Ht,
                               float* __restrict__ outv,
                               float* __restrict__ outi) {
  extern __shared__ float sc[];             // [16][SROW] score tile (padded)
  const int lane  = threadIdx.x & 31;
  const int wave  = threadIdx.x >> 5;
  const int b     = blockIdx.y;
  const int sbase = blockIdx.x * 16;
  const float NEG = -__builtin_inff();

  const __bf16* HsB = Hs + ((size_t)b * NN + sbase) * COUT;
  const __bf16* HtB = Ht + (size_t)b * NN * COUT;

  // ---- phase 1: WMMA score tile into LDS -------------------------------
  v16bf A[8];                               // 16x256 source rows, 8 K-frags
#pragma unroll
  for (int f = 0; f < 8; ++f) A[f] = load_a_frag(HsB, COUT, lane, f * 32);

  for (int tb = wave; tb < NN / 16; tb += 8) {
    const __bf16* Tp = HtB + (size_t)tb * 16 * COUT;
    if (tb + 8 < NN / 16)                   // global_prefetch_b8 next tile
      __builtin_prefetch((const char*)(Tp + (size_t)8 * 16 * COUT) + lane * 64, 0, 1);
    v8f acc = {};
#pragma unroll
    for (int f = 0; f < 8; ++f) {
      v16bf Bf = load_b_frag(Tp, COUT, lane, f * 32);  // Ht row-major == B^T
      acc = wmma_bf16(A[f], Bf, acc);
    }
    const int col    = tb * 16 + (lane & 15);
    const int rowoff = (lane >> 4) * 8;
#pragma unroll
    for (int r = 0; r < 8; ++r) sc[(size_t)(rowoff + r) * SROW + col] = acc[r];
  }
  __syncthreads();

  // ---- phase 2: per-row top-32 + softmax (wave handles 2 rows) ---------
#pragma unroll
  for (int rr = 0; rr < 2; ++rr) {
    const int row = wave * 2 + rr;
    float* srow = sc + (size_t)row * SROW;

    // per-lane sorted top-4 of this lane's 64-element strided slice
    float t0 = NEG, t1 = NEG, t2 = NEG, t3 = NEG;
    int   i0 = 0x7fffffff, i1 = 0x7fffffff, i2 = 0x7fffffff, i3 = 0x7fffffff;
    for (int j = lane; j < NN; j += 32)
      ins4(srow[j], j, t0, i0, t1, i1, t2, i2, t3, i3);

    float kval = 0.f;
    int   kidx = 0;
    for (int k = 0; k < KK; ++k) {
      float bv = t0;                        // wave32 argmax over lane heads
      int   bi = i0;
#pragma unroll
      for (int off = 16; off >= 1; off >>= 1) {
        float ov = __shfl_xor(bv, off, 32);
        int   oi = __shfl_xor(bi, off, 32);
        if (betterf(ov, oi, bv, bi)) { bv = ov; bi = oi; }
      }
      if (lane == k) { kval = bv; kidx = bi; }
      if (t0 == bv && i0 == bi) {           // winner lane: mark + pop head
        srow[bi] = NEG;
        t0 = t1; i0 = i1; t1 = t2; i1 = i2; t2 = t3; i2 = i3;
        t3 = NEG; i3 = 0x7fffffff;
        if (t0 == NEG) {                    // rare: local list exhausted
          asm volatile("s_wait_dscnt 0" ::: "memory");
          for (int j = lane; j < NN; j += 32)   // selected entries are -inf
            ins4(srow[j], j, t0, i0, t1, i1, t2, i2, t3, i3);
        }
      }
      asm volatile("s_wait_dscnt 0" ::: "memory");  // marks visible to all
    }

    // softmax over the 32 selected scores (lane 0 holds the max)
    float m = __shfl(kval, 0, 32);
    float e = __expf(kval - m);
    float s = e;
#pragma unroll
    for (int off = 16; off >= 1; off >>= 1) s += __shfl_xor(s, off, 32);
    const size_t g = (size_t)b * NN + sbase + row;
    outv[g * KK + lane] = e / s;
    outi[g * KK + lane] = (float)kidx;
  }
}

// ---------------------------------------------------------------------------
extern "C" void kernel_launch(void* const* d_in, const int* in_sizes, int n_in,
                              void* d_out, int out_size, void* d_ws, size_t ws_size,
                              hipStream_t stream) {
  const float* x_s = (const float*)d_in[0];
  const float* x_t = (const float*)d_in[3];
  const float* W   = (const float*)d_in[6];
  // edge_index_* / batch_* are unused by this psi choice (reference ignores them)

  char* ws = (char*)d_ws;
  __bf16* xs_bf = (__bf16*)ws;  ws += (size_t)BN * CIN * 2;
  __bf16* xt_bf = (__bf16*)ws;  ws += (size_t)BN * CIN * 2;
  __bf16* wt_bf = (__bf16*)ws;  ws += (size_t)COUT * CIN * 2;
  __bf16* hs_bf = (__bf16*)ws;  ws += (size_t)BN * COUT * 2;
  __bf16* ht_bf = (__bf16*)ws;  ws += (size_t)BN * COUT * 2;

  float* outv = (float*)d_out;
  float* outi = outv + (size_t)BN * KK;

  const int nconv = BN * CIN;
  cvt_bf16_kernel<<<2048, 256, 0, stream>>>(x_s, xs_bf, nconv);
  cvt_bf16_kernel<<<2048, 256, 0, stream>>>(x_t, xt_bf, nconv);
  wtrans_kernel<<<(CIN * COUT + 255) / 256, 256, 0, stream>>>(W, wt_bf);

  encode_kernel<<<BN / 16, 256, 0, stream>>>(xs_bf, wt_bf, hs_bf);
  encode_kernel<<<BN / 16, 256, 0, stream>>>(xt_bf, wt_bf, ht_bf);

  const int smem = 16 * SROW * sizeof(float);  // 128.25 KB (WGP has 320 KB LDS)
  hipFuncSetAttribute((const void*)simtopk_kernel,
                      hipFuncAttributeMaxDynamicSharedMemorySize, smem);
  simtopk_kernel<<<dim3(NN / 16, BB), 256, smem, stream>>>(hs_bf, ht_bf, outv, outi);
}
